// TopKSAE_49838800503342
// MI455X (gfx1250) — matrix operand order repaired and checked by reading, not compiled, not run on hardware
//
#include <hip/hip_runtime.h>
#include <hip/hip_bf16.h>

// ---------------------------------------------------------------------------
// TopK SAE forward for MI455X (gfx1250, wave32, WMMA, async-LDS staging).
//   BATCH=4096, D_INPUT=2048, N_LATENTS=32768, K=64
// Outputs (concat in d_out): x_hat (B*D), h (B*L), recon_loss (1), l0 (1)
// Workspace: W_enc as bf16, transposed [L][D]  (128 MB, L2-resident on MI455X)
//            topk list uint2[B][K]             (2 MB)
// ---------------------------------------------------------------------------

#define BATCH     4096
#define D_INPUT   2048
#define N_LATENTS 32768
#define TOPK      64

typedef __attribute__((ext_vector_type(16))) __bf16 bf16x16;
typedef __attribute__((ext_vector_type(8)))  float  f32x8;

union FragU { bf16x16 v; uint4 q[2]; };

// ---------------------------------------------------------------------------
// Kernel 1: W_enc f32 [D][L] -> bf16 transposed [L][D], via LDS tile transpose
// ---------------------------------------------------------------------------
__global__ __launch_bounds__(256)
void convert_transpose_kernel(const float* __restrict__ W, __bf16* __restrict__ wT)
{
    __shared__ __bf16 tile[32][36];              // +4 pad vs bank conflicts
    const int nBase = blockIdx.x * 32;           // latent dim
    const int kBase = blockIdx.y * 32;           // input dim
    const int t = threadIdx.x;

    // read: [k][n] coalesced (float4)
    {
        const int kk = t >> 3;                   // 0..31
        const int nn = (t & 7) * 4;              // 0..28
        float4 f = *(const float4*)(W + (size_t)(kBase + kk) * N_LATENTS + nBase + nn);
        tile[kk][nn + 0] = (__bf16)f.x;
        tile[kk][nn + 1] = (__bf16)f.y;
        tile[kk][nn + 2] = (__bf16)f.z;
        tile[kk][nn + 3] = (__bf16)f.w;
    }
    __syncthreads();
    // write: [n][k] coalesced (4 bf16 = 8B per thread)
    {
        const int on = t >> 3;                   // 0..31
        const int ok = (t & 7) * 4;              // 0..28
        union { __bf16 h[4]; uint2 q; } u;
        u.h[0] = tile[ok + 0][on];
        u.h[1] = tile[ok + 1][on];
        u.h[2] = tile[ok + 2][on];
        u.h[3] = tile[ok + 3][on];
        *(uint2*)(wT + (size_t)(nBase + on) * D_INPUT + kBase + ok) = u.q;
    }
}

// ---------------------------------------------------------------------------
// Kernel 2: h = ReLU((x - b_pre) @ W_enc), bf16 WMMA, f32 accumulate.
// 128x128 C-tile per block (8 waves: 4(M) x 2(N), each wave 32x64 = 8 accums).
// Double-buffered LDS; B tile staged with GLOBAL_LOAD_ASYNC_TO_LDS_B128
// (ASYNCcnt), A tile staged through VGPRs (needs f32->bf16 + b_pre fold).
// ---------------------------------------------------------------------------
#define TILE_M 128
#define TILE_N 128
#define TILE_K 32
#define LDSK   40   // padded LDS row length in bf16 units (32 data + 8 pad)
#define NK     (D_INPUT / TILE_K)   // 64 k-steps

__global__ __launch_bounds__(256)
void encode_gemm_kernel(const float* __restrict__ x, const __bf16* __restrict__ wT,
                        const float* __restrict__ b_pre, float* __restrict__ h)
{
    __shared__ __bf16 aT[2][TILE_M * LDSK];   // [m][k]  (x - b_pre) in bf16
    __shared__ __bf16 bT[2][TILE_N * LDSK];   // [n][k]  W_enc^T in bf16

    const int t     = threadIdx.x;
    const int mBase = blockIdx.y * TILE_M;
    const int nBase = blockIdx.x * TILE_N;

    const int w    = t >> 5;        // wave 0..7
    const int wm   = w >> 1;        // 0..3 (M)
    const int wn   = w & 1;         // 0..1 (N)
    const int lane = t & 31;
    const int lrow = lane & 15;
    const int lhi  = lane >> 4;

    const int lr = t >> 1;          // 0..127: row (A) / n (B) staged by this thread
    const int lh = t & 1;           // which 16-element half of the 32-wide k tile

    // Uniform base for async B staging (SGPR) + per-thread 32-bit byte offset.
    const unsigned long long wbase = (unsigned long long)wT;
    const unsigned bgoff0 = (((unsigned)(nBase + lr)) * (unsigned)D_INPUT
                             + (unsigned)(lh * 16)) * 2u;

    f32x8 acc[2][4];
#pragma unroll
    for (int i = 0; i < 2; ++i)
#pragma unroll
        for (int j = 0; j < 4; ++j)
#pragma unroll
            for (int r = 0; r < 8; ++r) acc[i][j][r] = 0.0f;

    // ---- staging helpers -------------------------------------------------
    auto stageA = [&](int buf, int k0) {
        const float* xp = x     + (size_t)(mBase + lr) * D_INPUT + k0 + lh * 16;
        const float* bp = b_pre + k0 + lh * 16;
        union { __bf16 hv[16]; uint4 q[2]; } u;
#pragma unroll
        for (int i = 0; i < 4; ++i) {
            float4 fx = ((const float4*)xp)[i];
            float4 fb = ((const float4*)bp)[i];
            u.hv[i * 4 + 0] = (__bf16)(fx.x - fb.x);
            u.hv[i * 4 + 1] = (__bf16)(fx.y - fb.y);
            u.hv[i * 4 + 2] = (__bf16)(fx.z - fb.z);
            u.hv[i * 4 + 3] = (__bf16)(fx.w - fb.w);
        }
        uint4* dst = (uint4*)(&aT[buf][lr * LDSK + lh * 16]);
        dst[0] = u.q[0];
        dst[1] = u.q[1];
    };
    // Async copy 32B of W_enc^T into LDS: two B128 transfers, ASYNCcnt-tracked.
    // offset:16 advances BOTH the LDS and global addresses (ISA 8 / §15.18.3).
    auto stageB = [&](int buf, unsigned goff) {
        unsigned ldsb = (unsigned)(uintptr_t)(&bT[buf][lr * LDSK + lh * 16]);
        asm volatile(
            "global_load_async_to_lds_b128 %0, %1, %2\n\t"
            "global_load_async_to_lds_b128 %0, %1, %2 offset:16"
            :: "v"(ldsb), "v"(goff), "s"(wbase) : "memory");
    };

    // ---- prologue: stage buffer 0 ---------------------------------------
    stageA(0, 0);
    stageB(0, bgoff0);

    for (int kt = 0; kt < NK; ++kt) {
        const int cur = kt & 1;
        const int nxt = cur ^ 1;
        const int k0n = (kt + 1) * TILE_K;

        if (kt + 1 < NK) {
            __builtin_prefetch(x + (size_t)(mBase + lr) * D_INPUT + k0n + TILE_K + lh * 16, 0, 1);
            stageA(nxt, k0n);
            stageB(nxt, bgoff0 + (unsigned)(k0n * 2));
            // ASYNCcnt is in-order: <=2 outstanding => current buffer's 2
            // transfers have landed; next buffer's 2 stay in flight.
            asm volatile("s_wait_asynccnt 0x2" ::: "memory");
        } else {
            asm volatile("s_wait_asynccnt 0x0" ::: "memory");
        }
        __syncthreads();

        // ---- build WMMA fragments per CDNA5 16-bit A/B VGPR layout ----
        // A (16x32): lanes 0-15 rows M=0..15; VGPR0-3 K=0..7 (lo)/8..15 (hi),
        //            VGPR4-7 K=16..23 (lo)/24..31 (hi).
        FragU afr[2];
#pragma unroll
        for (int i = 0; i < 2; ++i) {
            const __bf16* p = &aT[cur][(wm * 32 + i * 16 + lrow) * LDSK];
            afr[i].q[0] = *(const uint4*)(p + lhi * 8);
            afr[i].q[1] = *(const uint4*)(p + 16 + lhi * 8);
        }
        // B (32x16): lanes 0-15 column N=lane, K=0..7,16..23; lanes 16-31 K=8..15,24..31.
        FragU bfr[4];
#pragma unroll
        for (int j = 0; j < 4; ++j) {
            const __bf16* p = &bT[cur][(wn * 64 + j * 16 + lrow) * LDSK];
            bfr[j].q[0] = *(const uint4*)(p + lhi * 8);
            bfr[j].q[1] = *(const uint4*)(p + 16 + lhi * 8);
        }
#pragma unroll
        for (int i = 0; i < 2; ++i)
#pragma unroll
            for (int j = 0; j < 4; ++j)
                acc[i][j] = __builtin_amdgcn_wmma_f32_16x16x32_bf16(
                    false, afr[i].v, false, bfr[j].v, (short)0, acc[i][j], false, false);
        __syncthreads();   // protect buf[cur] before it is re-staged next iter
    }

    // ---- epilogue: ReLU + store. C layout: VGPR r = row r (lanes 0-15) / r+8 ----
#pragma unroll
    for (int i = 0; i < 2; ++i) {
#pragma unroll
        for (int j = 0; j < 4; ++j) {
            const int gcol = nBase + wn * 64 + j * 16 + lrow;
#pragma unroll
            for (int r = 0; r < 8; ++r) {
                const int grow = mBase + wm * 32 + i * 16 + r + 8 * lhi;
                float v = acc[i][j][r];
                h[(size_t)grow * N_LATENTS + gcol] = v > 0.0f ? v : 0.0f;
            }
        }
    }
}

// ---------------------------------------------------------------------------
// Kernel 3: per-row top-64 via float-bit histogram (values are >=0 post-ReLU,
// so the bit pattern is monotone). Zeroes non-selected h in place, emits
// compact (idx, val) list, accumulates l0.
// ---------------------------------------------------------------------------
#define NBINS 2048

__global__ __launch_bounds__(256)
void topk_kernel(float* __restrict__ h, uint2* __restrict__ list, float* __restrict__ l0_out)
{
    __shared__ unsigned hist[NBINS];
    __shared__ unsigned sThresh, sRem, sTakeTb, sNsel, sPos;

    const int row = blockIdx.x;
    float* hr = h + (size_t)row * N_LATENTS;
    uint2* lr = list + (size_t)row * TOPK;
    const int t = threadIdx.x;

    for (int i = t; i < NBINS; i += 256) hist[i] = 0u;
    if (t < TOPK) lr[t] = make_uint2(0u, 0u);   // pad so decode can read all 64
    __syncthreads();

    for (int i = t; i < N_LATENTS; i += 256) {
        unsigned bits = __float_as_uint(hr[i]);        // >= 0 -> sign bit clear
        atomicAdd(&hist[bits >> 20], 1u);              // 11-bit key: exp + 3 mantissa
    }
    __syncthreads();

    if (t == 0) {
        unsigned cum = 0;
        int b = NBINS - 1;
        for (; b > 0; --b) {
            if (cum + hist[b] >= TOPK) break;
            cum += hist[b];
        }
        sThresh = (unsigned)b;
        sRem    = TOPK - cum;      // quota taken from threshold bin
        sTakeTb = 0u; sNsel = 0u; sPos = 0u;
    }
    __syncthreads();

    const unsigned tb  = sThresh;
    const unsigned rem = sRem;
    for (int i = t; i < N_LATENTS; i += 256) {
        float v = hr[i];
        unsigned bin = __float_as_uint(v) >> 20;
        bool sel = false;
        if (bin > tb) {
            sel = true;
        } else if (bin == tb) {
            unsigned slot = atomicAdd(&sTakeTb, 1u);
            sel = slot < rem;
        }
        if (sel) {
            unsigned s = atomicAdd(&sNsel, 1u);
            if (s < TOPK) lr[s] = make_uint2((unsigned)i, __float_as_uint(v));
            if (v > 0.0f) atomicAdd(&sPos, 1u);
        } else {
            hr[i] = 0.0f;
        }
    }
    __syncthreads();
    if (t == 0) atomicAdd(l0_out, (float)sPos * (1.0f / (float)BATCH));
}

// ---------------------------------------------------------------------------
// Kernel 4: x_hat[row] = b_pre + sum_j val_j * W_dec[idx_j], 64 sparse rows.
// ---------------------------------------------------------------------------
__global__ __launch_bounds__(256)
void decode_kernel(const uint2* __restrict__ list, const float* __restrict__ Wdec,
                   const float* __restrict__ b_pre, float* __restrict__ xhat)
{
    __shared__ unsigned idxS[TOPK];
    __shared__ float    valS[TOPK];
    const int row = blockIdx.x;
    const int t = threadIdx.x;
    if (t < TOPK) {
        uint2 e = list[(size_t)row * TOPK + t];
        idxS[t] = e.x;
        valS[t] = __uint_as_float(e.y);
    }
    __syncthreads();

    const int c0 = t * 8;                 // 256 threads x 8 cols = 2048
    float4 a0 = *(const float4*)(b_pre + c0);
    float4 a1 = *(const float4*)(b_pre + c0 + 4);
    for (int j = 0; j < TOPK; ++j) {
        float v = valS[j];                // uniform -> scalar branch
        if (v != 0.0f) {
            const float* wr = Wdec + (size_t)idxS[j] * D_INPUT + c0;
            float4 w0 = *(const float4*)wr;
            float4 w1 = *(const float4*)(wr + 4);
            a0.x += v * w0.x; a0.y += v * w0.y; a0.z += v * w0.z; a0.w += v * w0.w;
            a1.x += v * w1.x; a1.y += v * w1.y; a1.z += v * w1.z; a1.w += v * w1.w;
        }
    }
    float* o = xhat + (size_t)row * D_INPUT + c0;
    *(float4*)o       = a0;
    *(float4*)(o + 4) = a1;
}

// ---------------------------------------------------------------------------
// Kernel 5: recon_loss = mean((x - x_hat)^2)
// ---------------------------------------------------------------------------
__global__ __launch_bounds__(256)
void loss_kernel(const float* __restrict__ x, const float* __restrict__ xhat,
                 float* __restrict__ out)
{
    __shared__ float red[256];
    const size_t N = (size_t)BATCH * D_INPUT;
    float s = 0.0f;
    for (size_t i = (size_t)blockIdx.x * 256 + threadIdx.x; i < N; i += (size_t)gridDim.x * 256) {
        float d = x[i] - xhat[i];
        s += d * d;
    }
    red[threadIdx.x] = s;
    __syncthreads();
    for (int o = 128; o > 0; o >>= 1) {
        if ((int)threadIdx.x < o) red[threadIdx.x] += red[threadIdx.x + o];
        __syncthreads();
    }
    if (threadIdx.x == 0)
        atomicAdd(out, red[0] * (1.0f / ((float)BATCH * (float)D_INPUT)));
}

// ---------------------------------------------------------------------------
extern "C" void kernel_launch(void* const* d_in, const int* in_sizes, int n_in,
                              void* d_out, int out_size, void* d_ws, size_t ws_size,
                              hipStream_t stream)
{
    (void)in_sizes; (void)n_in; (void)out_size; (void)ws_size;
    const float* x     = (const float*)d_in[0];
    const float* W_enc = (const float*)d_in[1];
    const float* W_dec = (const float*)d_in[2];
    const float* b_pre = (const float*)d_in[3];

    float* out   = (float*)d_out;
    float* xhat  = out;                                              // B*D
    float* h     = out + (size_t)BATCH * D_INPUT;                    // B*L
    float* recon = h   + (size_t)BATCH * N_LATENTS;                  // 1
    // l0 is recon + 1

    __bf16* wT  = (__bf16*)d_ws;                                     // [L][D] bf16
    uint2* list = (uint2*)((char*)d_ws + (size_t)N_LATENTS * D_INPUT * 2);

    hipMemsetAsync(recon, 0, 2 * sizeof(float), stream);             // recon + l0

    convert_transpose_kernel<<<dim3(N_LATENTS / 32, D_INPUT / 32), 256, 0, stream>>>(W_enc, wT);
    encode_gemm_kernel<<<dim3(N_LATENTS / TILE_N, BATCH / TILE_M), 256, 0, stream>>>(x, wT, b_pre, h);
    topk_kernel<<<BATCH, 256, 0, stream>>>(h, list, recon + 1);
    decode_kernel<<<BATCH, 256, 0, stream>>>(list, W_dec, b_pre, xhat);
    loss_kernel<<<2048, 256, 0, stream>>>(x, xhat, recon);
}
